// PLElayers_28595892257678
// MI455X (gfx1250) — matrix operand order repaired and checked by reading, not compiled, not run on hardware
//
#include <hip/hip_runtime.h>
#include <stdint.h>

// ---------------------------------------------------------------------------
// Fused PLE / multi-gate MoE for MI455X (gfx1250, wave32, WMMA bf16)
//   experts: [B,256] x [8,256,128] (+bias) for 3 banks
//   gates:   softmax(x @ G^T) weighted sums over concatenated expert banks
// ---------------------------------------------------------------------------

typedef __attribute__((ext_vector_type(16))) __bf16 v16bf;
typedef __attribute__((ext_vector_type(8)))  float  v8f;

#define BATCH    16384
#define IN_DIM   256
#define EXP_DIM  128
#define NEXP     8
#define RT       2                 // row-tiles (16 rows each) per block
#define ROWS_PER_BLK (16*RT)

// packed B-operand fragment: 32 lanes x 16 bf16 = 512 elems = 1KB
#define FRAG_ELEMS 512
#define WPK_ELEMS (3*NEXP*8*8*FRAG_ELEMS)   // banks*experts*ntiles*kblocks
#define GPK_ELEMS (4*8*FRAG_ELEMS)          // 4 gate tiles * 8 kblocks

// LDS layout (dynamic shared)
#define STAGE_BYTES (3*ROWS_PER_BLK*IN_DIM*4)       // 98304: fp32 x tiles
#define AFRAG_BYTES (RT*3*8*FRAG_ELEMS*2)           // 49152: bf16 A fragments
#define WGT_BYTES   (RT*3*16*32*4)                  // 12288: softmax weights
#define SMEM_BYTES  (STAGE_BYTES + AFRAG_BYTES + WGT_BYTES)

__device__ __forceinline__ unsigned short f2bf(float f){
  union { float f; unsigned u; } c; c.f = f;
  unsigned u = c.u;
  return (unsigned short)((u + 0x7FFFu + ((u >> 16) & 1u)) >> 16);  // RNE
}

__device__ __forceinline__ v8f wmma_bf16(v16bf a, v16bf b, v8f c){
  return __builtin_amdgcn_wmma_f32_16x16x32_bf16(false, a, false, b,
                                                 (short)0, c, false, false);
}

// ---------------------------------------------------------------------------
// Pack expert weights [E,IN,D] fp32 -> bf16 WMMA B-operand fragments.
// Source-coalesced: thread walks (bank,e,k,col4), reads float4 along col.
// B layout (16-bit, 32x16): lane = n + 16*(K-half), slot = K&15.
// Fragment order: ((bank*8+e)*8+nt)*8+kb, then lane*16+slot.
// ---------------------------------------------------------------------------
__global__ void pack_w_kernel(const float* __restrict__ Wa,
                              const float* __restrict__ Wb,
                              const float* __restrict__ Ws,
                              unsigned short* __restrict__ out){
  int t = blockIdx.x * 256 + threadIdx.x;      // one float4 (4 cols) per thread
  int col0 = (t & 31) << 2;                    // 0..124, step 4
  int k    = (t >> 5) & 255;
  int e    = (t >> 13) & 7;
  int bank = t >> 16;
  const float* W = (bank == 0) ? Wa : (bank == 1) ? Wb : Ws;
  const float4 v = *(const float4*)(W + (e * IN_DIM + k) * EXP_DIM + col0);
  int kb   = k >> 5;
  int kin  = k & 31;
  int half = kin >> 4;
  int slot = kin & 15;
  float vv[4] = { v.x, v.y, v.z, v.w };
#pragma unroll
  for (int c = 0; c < 4; ++c){
    int col  = col0 + c;
    int nt   = col >> 4;
    int lane = (col & 15) + (half << 4);
    out[(((bank * 8 + e) * 8 + nt) * 8 + kb) * FRAG_ELEMS + lane * 16 + slot] = f2bf(vv[c]);
  }
}

// Gate matrices -> B fragments. tiles: 0=Ga(16), 1=Gb(16), 2=Gs[0:16], 3=Gs[16:24]+pad
__global__ void pack_g_kernel(const float* __restrict__ Ga,
                              const float* __restrict__ Gb,
                              const float* __restrict__ Gs,
                              unsigned short* __restrict__ out){
  int t  = blockIdx.x * 256 + threadIdx.x;     // one float4 (4 k) per thread
  int k0 = (t & 63) << 2;                      // 0..252, step 4
  int n  = (t >> 6) & 15;
  int tt = t >> 10;                            // tile 0..3
  float4 v = make_float4(0.f, 0.f, 0.f, 0.f);
  if      (tt == 0) v = *(const float4*)(Ga + n * IN_DIM + k0);
  else if (tt == 1) v = *(const float4*)(Gb + n * IN_DIM + k0);
  else if (tt == 2) v = *(const float4*)(Gs + n * IN_DIM + k0);
  else if (n < 8)   v = *(const float4*)(Gs + (16 + n) * IN_DIM + k0);
  int kb   = k0 >> 5;
  int kin  = k0 & 31;
  int half = kin >> 4;
  int slot = kin & 15;                         // 4-aligned
  int lane = n + (half << 4);
  ushort4 o;
  o.x = f2bf(v.x); o.y = f2bf(v.y); o.z = f2bf(v.z); o.w = f2bf(v.w);
  *(ushort4*)(out + (tt * 8 + kb) * FRAG_ELEMS + lane * 16 + slot) = o;
}

// ---------------------------------------------------------------------------
// Main fused kernel: 256 threads (8 waves), 32 rows per block, wave = N-tile.
// ---------------------------------------------------------------------------
__global__ __launch_bounds__(256, 1)
void moe_fused_kernel(const float* __restrict__ xa,
                      const float* __restrict__ xb,
                      const float* __restrict__ xs,
                      const float* __restrict__ ba,
                      const float* __restrict__ bb,
                      const float* __restrict__ bs,
                      const unsigned short* __restrict__ wpk,
                      const unsigned short* __restrict__ gpk,
                      float* __restrict__ out){
  extern __shared__ char smem[];
  float*          stage = (float*)smem;                          // fp32 x tiles
  unsigned short* aFrag = (unsigned short*)(smem + STAGE_BYTES); // bf16 A frags
  float*          wgt   = (float*)(smem + STAGE_BYTES + AFRAG_BYTES);

  const int tid  = threadIdx.x;
  const int lane = tid & 31;
  const int wave = __builtin_amdgcn_readfirstlane(tid >> 5);  // scalar wave id
  const int row0 = blockIdx.x * ROWS_PER_BLK;
  const int h    = lane >> 4;       // K/M half select
  const int n    = lane & 15;

  // ---- stage x tiles: 3 contiguous 32KB regions, async global->LDS ----
  {
    const float* srcs[3] = { xa + (size_t)row0 * IN_DIM,
                             xb + (size_t)row0 * IN_DIM,
                             xs + (size_t)row0 * IN_DIM };
#pragma unroll
    for (int bnk = 0; bnk < 3; ++bnk){
      uint32_t ldsbase = (uint32_t)(uintptr_t)(stage + bnk * ROWS_PER_BLK * IN_DIM);
      const char* g = (const char*)srcs[bnk];
#pragma unroll
      for (int i = 0; i < 8; ++i){
        uint32_t off    = (uint32_t)(tid * 16 + i * 4096);
        uint32_t ldsoff = ldsbase + off;
        uint64_t gaddr  = (uint64_t)(uintptr_t)(g + off);
        asm volatile("global_load_async_to_lds_b128 %0, %1, off"
                     :: "v"(ldsoff), "v"(gaddr) : "memory");
      }
    }
    asm volatile("s_wait_asynccnt 0x0" ::: "memory");
  }
  __syncthreads();

  // ---- convert fp32 -> bf16 A fragments (contiguous-K A layout, same K order as packed B), 4 at a time ----
  // A 16x32: lane = m + 16*hh; K-in-block: hh0:{0-15}, hh1:{16-31}
#pragma unroll
  for (int j = 0; j < (3 * ROWS_PER_BLK * IN_DIM) / (256 * 4); ++j){
    int i   = (tid + j * 256) << 2;       // element index, 4-aligned
    int bnk = i >> 13;
    int r   = i & 8191;
    int m   = r >> 8;
    int k   = r & 255;
    int rt  = m >> 4;
    int mm  = m & 15;
    int kb  = k >> 5;
    int k2  = k & 31;
    int grp  = k2 >> 3;
    int hh   = k2 >> 4;
    int slot = k2 & 15;                   // 4-aligned
    int ln   = mm + (hh << 4);
    const float4 v = *(const float4*)(stage + bnk * (ROWS_PER_BLK * IN_DIM) + m * IN_DIM + k);
    ushort4 o;
    o.x = f2bf(v.x); o.y = f2bf(v.y); o.z = f2bf(v.z); o.w = f2bf(v.w);
    *(ushort4*)(aFrag + ((rt * 3 + bnk) * 8 + kb) * FRAG_ELEMS + ln * 16 + slot) = o;
  }
  __syncthreads();

  // ---- gate phase: waves 0..5 -> (row-tile, gate) logits + softmax ----
  if (wave < 3 * RT){                       // scalar branch
    const int rt = wave / 3;
    const int g  = wave % 3;
    const unsigned short* abase = aFrag + (rt * 3 + g) * 8 * FRAG_ELEMS + lane * 16;
    v8f c0 = {}, c1 = {};
    if (g < 2){                             // scalar branch: single 16-col tile
      const unsigned short* gbase = gpk + g * 8 * FRAG_ELEMS + lane * 16;
#pragma unroll
      for (int kb = 0; kb < 8; ++kb){
        const v16bf a  = *(const v16bf*)(abase + kb * FRAG_ELEMS);
        const v16bf b0 = *(const v16bf*)(gbase + kb * FRAG_ELEMS);
        c0 = wmma_bf16(a, b0, c0);
      }
#pragma unroll
      for (int q = 0; q < 8; ++q){
        float e0 = c0[q];
        float mx = e0;
        for (int m = 8; m >= 1; m >>= 1) mx = fmaxf(mx, __shfl_xor(mx, m, 16));
        float x0 = __expf(e0 - mx);
        float s  = x0;
        for (int m = 8; m >= 1; m >>= 1) s += __shfl_xor(s, m, 16);
        wgt[((rt * 3 + g) * 16 + q + (h << 3)) * 32 + n] = x0 / s;
      }
    } else {                                // shared gate: 24 logits, 2 tiles
      const unsigned short* g0 = gpk + 2 * 8 * FRAG_ELEMS + lane * 16;
      const unsigned short* g1 = gpk + 3 * 8 * FRAG_ELEMS + lane * 16;
#pragma unroll
      for (int kb = 0; kb < 8; ++kb){
        const v16bf a  = *(const v16bf*)(abase + kb * FRAG_ELEMS);
        const v16bf b0 = *(const v16bf*)(g0 + kb * FRAG_ELEMS);
        const v16bf b1 = *(const v16bf*)(g1 + kb * FRAG_ELEMS);
        c0 = wmma_bf16(a, b0, c0);
        c1 = wmma_bf16(a, b1, c1);
      }
      bool t1v = (n < 8);                   // cols 16..23 valid in tile 1
#pragma unroll
      for (int q = 0; q < 8; ++q){
        float e0 = c0[q];
        float e1 = t1v ? c1[q] : -3.0e38f;
        float mx = fmaxf(e0, e1);
        for (int m = 8; m >= 1; m >>= 1) mx = fmaxf(mx, __shfl_xor(mx, m, 16));
        float x0 = __expf(e0 - mx);
        float x1 = t1v ? __expf(e1 - mx) : 0.f;
        float s  = x0 + x1;
        for (int m = 8; m >= 1; m >>= 1) s += __shfl_xor(s, m, 16);
        float inv = 1.f / s;
        int row = q + (h << 3);
        wgt[((rt * 3 + g) * 16 + row) * 32 + n] = x0 * inv;
        if (t1v) wgt[((rt * 3 + g) * 16 + row) * 32 + 16 + n] = x1 * inv;
      }
    }
  }
  __syncthreads();

  // ---- expert phase: wave owns N-tile nt; 3 banks x 8 experts x 8 K-steps ----
  const int nt  = wave;                     // scalar
  const int col = nt * 16 + n;
  v8f oA0 = {}, oA1 = {}, oB0 = {}, oB1 = {}, oS0 = {}, oS1 = {};

  for (int bnk = 0; bnk < 3; ++bnk){
    const float* bp = (bnk == 0) ? ba : (bnk == 1) ? bb : bs;
    const unsigned short* a0base = aFrag + (0 * 3 + bnk) * 8 * FRAG_ELEMS + lane * 16;
    const unsigned short* a1base = aFrag + (1 * 3 + bnk) * 8 * FRAG_ELEMS + lane * 16;
    const unsigned short* wbase  = wpk + ((bnk * 8) * 8 + nt) * 8 * FRAG_ELEMS + lane * 16;
#pragma unroll
    for (int e = 0; e < NEXP; ++e){
      float bv = bp[e * EXP_DIM + col];
      v8f a0, a1;
#pragma unroll
      for (int q = 0; q < 8; ++q){ a0[q] = bv; a1[q] = bv; }
      const unsigned short* wb = wbase + (e * 8 * 8) * FRAG_ELEMS;
#pragma unroll
      for (int kb = 0; kb < 8; ++kb){
        const v16bf bf  = *(const v16bf*)(wb + kb * FRAG_ELEMS);
        const v16bf af0 = *(const v16bf*)(a0base + kb * FRAG_ELEMS);
        a0 = wmma_bf16(af0, bf, a0);
        const v16bf af1 = *(const v16bf*)(a1base + kb * FRAG_ELEMS);
        a1 = wmma_bf16(af1, bf, a1);
      }
#pragma unroll
      for (int rt = 0; rt < RT; ++rt){
        v8f& acc = rt ? a1 : a0;
        v8f& oA  = rt ? oA1 : oA0;
        v8f& oB  = rt ? oB1 : oB0;
        v8f& oS  = rt ? oS1 : oS0;
#pragma unroll
        for (int q = 0; q < 8; ++q){
          int row = q + (h << 3);
          float t = acc[q];
          const float* w0 = &wgt[((rt * 3 + 0) * 16 + row) * 32];
          const float* w1 = &wgt[((rt * 3 + 1) * 16 + row) * 32];
          const float* w2 = &wgt[((rt * 3 + 2) * 16 + row) * 32];
          if (bnk == 0){        // ea feeds out_a (idx e) and out_s (idx e)
            oA[q] += w0[e] * t;
            oS[q] += w2[e] * t;
          } else if (bnk == 1){ // eb feeds out_b (idx e) and out_s (idx 8+e)
            oB[q] += w1[e] * t;
            oS[q] += w2[8 + e] * t;
          } else {              // es feeds out_a/out_b (idx 8+e), out_s (16+e)
            oA[q] += w0[8 + e] * t;
            oB[q] += w1[8 + e] * t;
            oS[q] += w2[16 + e] * t;
          }
        }
      }
    }
  }

  // ---- store: out = [out_a | out_b | out_s], each [B, 128] fp32 ----
  float* outA = out;
  float* outB = out + (size_t)BATCH * EXP_DIM;
  float* outS = out + 2 * (size_t)BATCH * EXP_DIM;
#pragma unroll
  for (int rt = 0; rt < RT; ++rt){
    const v8f& oA = rt ? oA1 : oA0;
    const v8f& oB = rt ? oB1 : oB0;
    const v8f& oS = rt ? oS1 : oS0;
#pragma unroll
    for (int q = 0; q < 8; ++q){
      size_t row = (size_t)row0 + rt * 16 + q + (h << 3);
      outA[row * EXP_DIM + col] = oA[q];
      outB[row * EXP_DIM + col] = oB[q];
      outS[row * EXP_DIM + col] = oS[q];
    }
  }
}

// ---------------------------------------------------------------------------
extern "C" void kernel_launch(void* const* d_in, const int* in_sizes, int n_in,
                              void* d_out, int out_size, void* d_ws, size_t ws_size,
                              hipStream_t stream) {
  (void)in_sizes; (void)n_in; (void)out_size; (void)ws_size;
  const float* xa = (const float*)d_in[0];
  const float* xb = (const float*)d_in[1];
  const float* xs = (const float*)d_in[2];
  const float* Wa = (const float*)d_in[3];
  const float* ba = (const float*)d_in[4];
  const float* Wb = (const float*)d_in[5];
  const float* bb = (const float*)d_in[6];
  const float* Ws = (const float*)d_in[7];
  const float* bs = (const float*)d_in[8];
  const float* Ga = (const float*)d_in[9];
  const float* Gb = (const float*)d_in[10];
  const float* Gs = (const float*)d_in[11];

  unsigned short* wpk = (unsigned short*)d_ws;          // 1.5 MB packed weights
  unsigned short* gpk = wpk + WPK_ELEMS;                // 32 KB packed gates

  pack_w_kernel<<<(WPK_ELEMS / 4) / 256, 256, 0, stream>>>(Wa, Wb, Ws, wpk);
  pack_g_kernel<<<(GPK_ELEMS / 4) / 256, 256, 0, stream>>>(Ga, Gb, Gs, gpk);

  moe_fused_kernel<<<BATCH / ROWS_PER_BLK, 256, SMEM_BYTES, stream>>>(
      xa, xb, xs, ba, bb, bs, wpk, gpk, (float*)d_out);
}
